// TemporalMemoryWithSharedMLP_58806692217150
// MI455X (gfx1250) — matrix operand order repaired and checked by reading, not compiled, not run on hardware
//
#include <hip/hip_runtime.h>
#include <hip/hip_bf16.h>
#include <math.h>

typedef __attribute__((ext_vector_type(16))) __bf16 v16bf;
typedef __attribute__((ext_vector_type(8)))  float  v8f;

#define T_TOK   16384
#define C_DIM   512
#define MEMN    2000
#define JPAD    2016     // MEM padded to multiple of 32 for WMMA K
#define HWD     1024
#define N_IMG   16
#define EPSV    1e-12f
#define SHRINKV 0.0025f

__device__ __forceinline__ unsigned short f2bf(float f) {
  unsigned int u = __float_as_uint(f);
  u += 0x7FFFu + ((u >> 16) & 1u);   // round-to-nearest-even
  return (unsigned short)(u >> 16);
}

union Frag {
  v16bf v;
  uint4 q[2];
  unsigned short s[16];
};

// CDNA5 async global->LDS copy (ASYNCcnt-tracked), per-lane b128.
// lds_off: wave-relative LDS byte address (low 32 bits of the flat pointer).
__device__ __forceinline__ void async_copy_b128(unsigned int lds_off, const void* gptr) {
  asm volatile("global_load_async_to_lds_b128 %0, %1, off"
               :: "v"(lds_off), "v"((unsigned long long)(uintptr_t)gptr)
               : "memory");
}
__device__ __forceinline__ void wait_async0() {
  asm volatile("s_wait_asynccnt 0x0" ::: "memory");
}

// ---------------------------------------------------------------------------
// Kernel 1: q = l2norm over C of x (NCHW -> token-major), packed to bf16.
// One wave per token; lane covers c = lane + 32*i.
// ---------------------------------------------------------------------------
__global__ void qnorm_kernel(const float* __restrict__ x,
                             unsigned short* __restrict__ qbf) {
  const int wave = threadIdx.x >> 5;
  const int lane = threadIdx.x & 31;
  const int t    = blockIdx.x * 4 + wave;
  const int nimg = t >> 10;
  const int hw   = t & (HWD - 1);
  const float* xb = x + (size_t)nimg * C_DIM * HWD + hw;
  float vals[16];
  float ss = 0.f;
#pragma unroll
  for (int i = 0; i < 16; ++i) {
    const int c = lane + i * 32;
    float v = xb[(size_t)c * HWD];
    vals[i] = v;
    ss += v * v;
  }
#pragma unroll
  for (int m = 16; m >= 1; m >>= 1) ss += __shfl_xor(ss, m, 32);
  const float scale = 1.0f / fmaxf(sqrtf(ss), EPSV);
  unsigned short* qrow = qbf + (size_t)t * C_DIM;
#pragma unroll
  for (int i = 0; i < 16; ++i) {
    const int c = lane + i * 32;
    qrow[c] = f2bf(vals[i] * scale);
  }
}

// ---------------------------------------------------------------------------
// Kernel 2: memory MLP (Linear-ReLU-Linear-ReLU) + row L2 norm.
// One 256-thread block per memory row. Writes m_norm in bf16 twice:
//   mbf  [JPAD][C_DIM]  (row-major, K along C)  -> B of GEMM1
//   mbft [C_DIM][JPAD]  (transposed, K along j) -> B of GEMM2
// ---------------------------------------------------------------------------
__global__ void mlp_kernel(const float* __restrict__ memory,
                           const float* __restrict__ w1,
                           const float* __restrict__ b1,
                           const float* __restrict__ w2,
                           const float* __restrict__ b2,
                           unsigned short* __restrict__ mbf,
                           unsigned short* __restrict__ mbft) {
  __shared__ float mrow[C_DIM];
  __shared__ float hbuf[256];
  __shared__ float red[8];
  __shared__ float snorm;
  const int row = blockIdx.x;
  const int tid = threadIdx.x;
  mrow[tid]       = memory[(size_t)row * C_DIM + tid];
  mrow[tid + 256] = memory[(size_t)row * C_DIM + tid + 256];
  __syncthreads();
  {
    const float* wr = w1 + (size_t)tid * C_DIM;
    float acc = b1[tid];
    for (int k = 0; k < C_DIM; ++k) acc += mrow[k] * wr[k];
    hbuf[tid] = fmaxf(acc, 0.f);
  }
  __syncthreads();
  float m0, m1;
  {
    const float* wr0 = w2 + (size_t)tid * 256;
    const float* wr1 = w2 + (size_t)(tid + 256) * 256;
    float a0 = b2[tid], a1 = b2[tid + 256];
    for (int k = 0; k < 256; ++k) { const float h = hbuf[k]; a0 += h * wr0[k]; a1 += h * wr1[k]; }
    m0 = fmaxf(a0, 0.f);
    m1 = fmaxf(a1, 0.f);
  }
  float ss = m0 * m0 + m1 * m1;
#pragma unroll
  for (int msk = 16; msk >= 1; msk >>= 1) ss += __shfl_xor(ss, msk, 32);
  if ((tid & 31) == 0) red[tid >> 5] = ss;
  __syncthreads();
  if (tid == 0) {
    float tot = 0.f;
    for (int i = 0; i < 8; ++i) tot += red[i];
    snorm = 1.0f / fmaxf(sqrtf(tot), EPSV);
  }
  __syncthreads();
  const float sc = snorm;
  const unsigned short q0 = f2bf(m0 * sc);
  const unsigned short q1 = f2bf(m1 * sc);
  mbf[(size_t)row * C_DIM + tid]            = q0;
  mbf[(size_t)row * C_DIM + tid + 256]      = q1;
  mbft[(size_t)tid * JPAD + row]            = q0;
  mbft[(size_t)(tid + 256) * JPAD + row]    = q1;
}

// ---------------------------------------------------------------------------
// Kernel 3: fused attention. One block = 16 tokens, 4 waves.
// Phase A: GEMM1  S = q_tile @ m_norm^T   (bf16 WMMA, f32 accum, S in LDS)
// Phase B: softmax + shrink + L1 renorm in LDS, coalesced att store, bf16 pack
// Phase C: GEMM2  out = att @ m_norm      (bf16 WMMA), staged coalesced store
// ---------------------------------------------------------------------------
__global__ void __launch_bounds__(128)
attn_kernel(const unsigned short* __restrict__ qbf,
            const unsigned short* __restrict__ mbf,
            const unsigned short* __restrict__ mbft,
            float* __restrict__ out,
            float* __restrict__ attout) {
  extern __shared__ char smem[];
  float*          Sld   = (float*)smem;                                   // 16 x JPAD f32
  unsigned short* qtile = (unsigned short*)(smem + 16 * JPAD * 4);        // 16 x 512 bf16
  unsigned short* attbf = qtile + 16 * C_DIM;                             // 16 x JPAD bf16

  const int tid    = threadIdx.x;
  const int wave   = tid >> 5;
  const int lane   = tid & 31;
  const int nlo    = lane & 15;
  const int hi     = lane >> 4;
  const int t_base = blockIdx.x * 16;
  const int nimg   = t_base >> 10;            // 16-token tile never crosses an image
  const int hwb    = t_base & (HWD - 1);

  // --- stage q tile (16 KB) via CDNA5 async global->LDS copies ---
  {
    const unsigned int qtile_off = (unsigned int)(uintptr_t)qtile;  // LDS byte offset
    const unsigned short* src = qbf + (size_t)t_base * C_DIM;
#pragma unroll
    for (int i = 0; i < 8; ++i) {
      const int idx = tid + i * 128;                 // 1024 x b128 total
      async_copy_b128(qtile_off + idx * 16, src + idx * 8);
    }
    wait_async0();
  }
  __syncthreads();

  // --- Phase A: GEMM1, j-tiles round-robined over the 4 waves ---
  for (int jt = wave; jt < MEMN / 16; jt += 4) {
    const int j0 = jt * 16;
    v8f acc0 = {}, acc1 = {};
    const unsigned short* brow = mbf + (size_t)(j0 + nlo) * C_DIM;  // B: row j, K along C
    const unsigned short* arow = qtile + nlo * C_DIM;               // A: token row, K along C
    if (jt + 4 < MEMN / 16)                                         // prefetch next B row
      __builtin_prefetch(mbf + (size_t)((jt + 4) * 16 + nlo) * C_DIM, 0, 0);
#pragma unroll 2
    for (int ks = 0; ks < C_DIM / 32; ks += 2) {
      const int k0 = ks * 32;
      {
        Frag a, b;
        a.q[0] = *(const uint4*)(arow + k0 + hi * 8);
        a.q[1] = *(const uint4*)(arow + k0 + 16 + hi * 8);
        b.q[0] = *(const uint4*)(brow + k0 + hi * 8);
        b.q[1] = *(const uint4*)(brow + k0 + 16 + hi * 8);
        acc0 = __builtin_amdgcn_wmma_f32_16x16x32_bf16(false, a.v, false, b.v,
                                                       (short)0, acc0, false, false);
      }
      {
        const int k1 = k0 + 32;
        Frag a, b;
        a.q[0] = *(const uint4*)(arow + k1 + hi * 8);
        a.q[1] = *(const uint4*)(arow + k1 + 16 + hi * 8);
        b.q[0] = *(const uint4*)(brow + k1 + hi * 8);
        b.q[1] = *(const uint4*)(brow + k1 + 16 + hi * 8);
        acc1 = __builtin_amdgcn_wmma_f32_16x16x32_bf16(false, a.v, false, b.v,
                                                       (short)0, acc1, false, false);
      }
    }
    const v8f acc = acc0 + acc1;
#pragma unroll
    for (int r = 0; r < 8; ++r) Sld[(r + 8 * hi) * JPAD + j0 + nlo] = acc[r];
  }
  __syncthreads();

  // --- Phase B: per-row softmax -> shrink -> L1 renorm (8 lanes per row) ---
  {
    const int row = tid >> 3;
    const int sub = tid & 7;
    float* srow = Sld + row * JPAD;
    float mx = -3.0e38f;
    for (int j = sub; j < MEMN; j += 8) mx = fmaxf(mx, srow[j]);
    mx = fmaxf(mx, __shfl_xor(mx, 1, 8));
    mx = fmaxf(mx, __shfl_xor(mx, 2, 8));
    mx = fmaxf(mx, __shfl_xor(mx, 4, 8));
    float z = 0.f;
    for (int j = sub; j < MEMN; j += 8) { const float p = __expf(srow[j] - mx); srow[j] = p; z += p; }
    z += __shfl_xor(z, 1, 8); z += __shfl_xor(z, 2, 8); z += __shfl_xor(z, 4, 8);
    const float iz = 1.0f / fmaxf(z, EPSV);
    float s1 = 0.f;
    for (int j = sub; j < MEMN; j += 8) { const float a = fmaxf(srow[j] * iz - SHRINKV, 0.f); srow[j] = a; s1 += a; }
    s1 += __shfl_xor(s1, 1, 8); s1 += __shfl_xor(s1, 2, 8); s1 += __shfl_xor(s1, 4, 8);
    const float il1 = 1.0f / fmaxf(s1, EPSV);
    unsigned short* abf = attbf + row * JPAD;
    for (int j = sub; j < MEMN; j += 8) {
      const float a = srow[j] * il1;
      srow[j] = a;
      abf[j]  = f2bf(a);
    }
    abf[MEMN + sub]     = 0;   // zero K padding
    abf[MEMN + 8 + sub] = 0;
  }
  __syncthreads();

  // --- coalesced att store: for fixed j, 16 tile tokens are contiguous ---
  {
    float* abase = attout + (size_t)nimg * MEMN * HWD + hwb;
    for (int idx = tid; idx < MEMN * 16; idx += 128) {
      const int j = idx >> 4;
      const int r = idx & 15;
      abase[(size_t)j * HWD + r] = Sld[r * JPAD + j];
    }
  }
  __syncthreads();   // Sld now reusable as output staging

  // --- Phase C: GEMM2, c-tiles round-robined over waves; K = JPAD over j ---
  float* stage = Sld + wave * 256;   // wave-private 16x16 f32 staging
  for (int cti = 0; cti < 8; ++cti) {
    const int c0 = (wave + cti * 4) * 16;
    v8f acc0 = {}, acc1 = {};
    const unsigned short* brow = mbft + (size_t)(c0 + nlo) * JPAD;  // B: col c, K along j
    const unsigned short* arow = attbf + nlo * JPAD;                // A: token row, K along j
    if (cti + 1 < 8)                                                // prefetch next B row
      __builtin_prefetch(mbft + (size_t)((wave + (cti + 1) * 4) * 16 + nlo) * JPAD, 0, 0);
#pragma unroll 2
    for (int ks = 0; ks < JPAD / 32; ks += 2) {
      const int k0 = ks * 32;
      {
        Frag a, b;
        a.q[0] = *(const uint4*)(arow + k0 + hi * 8);
        a.q[1] = *(const uint4*)(arow + k0 + 16 + hi * 8);
        b.q[0] = *(const uint4*)(brow + k0 + hi * 8);
        b.q[1] = *(const uint4*)(brow + k0 + 16 + hi * 8);
        acc0 = __builtin_amdgcn_wmma_f32_16x16x32_bf16(false, a.v, false, b.v,
                                                       (short)0, acc0, false, false);
      }
      if (ks + 1 < JPAD / 32) {
        const int k1 = k0 + 32;
        Frag a, b;
        a.q[0] = *(const uint4*)(arow + k1 + hi * 8);
        a.q[1] = *(const uint4*)(arow + k1 + 16 + hi * 8);
        b.q[0] = *(const uint4*)(brow + k1 + hi * 8);
        b.q[1] = *(const uint4*)(brow + k1 + 16 + hi * 8);
        acc1 = __builtin_amdgcn_wmma_f32_16x16x32_bf16(false, a.v, false, b.v,
                                                       (short)0, acc1, false, false);
      }
    }
    const v8f acc = acc0 + acc1;
#pragma unroll
    for (int r = 0; r < 8; ++r) stage[nlo * 16 + r + 8 * hi] = acc[r];
    // coalesced store: for fixed c, 16 tile tokens are contiguous
    float* obase = out + (size_t)nimg * C_DIM * HWD + hwb;
    for (int e = lane; e < 256; e += 32) {
      const int c = e >> 4;
      const int r = e & 15;
      obase[(size_t)(c0 + c) * HWD + r] = stage[e];
    }
  }
}

extern "C" void kernel_launch(void* const* d_in, const int* in_sizes, int n_in,
                              void* d_out, int out_size, void* d_ws, size_t ws_size,
                              hipStream_t stream) {
  (void)in_sizes; (void)n_in; (void)out_size; (void)ws_size;
  const float* x      = (const float*)d_in[0];
  const float* memory = (const float*)d_in[1];
  const float* w1     = (const float*)d_in[2];
  const float* b1     = (const float*)d_in[3];
  const float* w2     = (const float*)d_in[4];
  const float* b2     = (const float*)d_in[5];

  float* out    = (float*)d_out;
  float* attout = out + (size_t)N_IMG * C_DIM * HWD;   // outputs concatenated: output, att_r

  // workspace: qbf [T,512] bf16 | mbf [JPAD,512] bf16 | mbft [512,JPAD] bf16 (~21 MB)
  char* ws = (char*)d_ws;
  unsigned short* qbf  = (unsigned short*)ws;
  unsigned short* mbf  = (unsigned short*)(ws + (size_t)T_TOK * C_DIM * 2);
  unsigned short* mbft = mbf + (size_t)JPAD * C_DIM;

  // zero padded rows/cols of both m_norm copies (mbf & mbft are contiguous)
  hipMemsetAsync(mbf, 0, (size_t)JPAD * C_DIM * 2 * 2, stream);

  qnorm_kernel<<<T_TOK / 4, 128, 0, stream>>>(x, qbf);
  mlp_kernel<<<MEMN, 256, 0, stream>>>(memory, w1, b1, w2, b2, mbf, mbft);

  const size_t lds_bytes = (size_t)16 * JPAD * 4      // S / staging (f32)
                         + (size_t)16 * C_DIM * 2     // q tile (bf16)
                         + (size_t)16 * JPAD * 2;     // att tile (bf16)  ~205 KB (<320 KB/WG)
  attn_kernel<<<T_TOK / 16, 128, lds_bytes, stream>>>(qbf, mbf, mbft, out, attout);
}
